// M2A_4604204941664
// MI455X (gfx1250) — matrix operand order repaired and checked by reading, not compiled, not run on hardware
//
#include <hip/hip_runtime.h>

#define DCH 128
#define GN_EPS 1e-5f

typedef __attribute__((ext_vector_type(16))) __bf16 v16bf;
typedef __attribute__((ext_vector_type(8)))  float  v8f;
typedef __attribute__((ext_vector_type(4)))  unsigned int u32x4;

// ---------------- wave-local LDS fence (gfx1250 split counters) ----------------
__device__ __forceinline__ void lds_fence_wave() {
    __builtin_amdgcn_wave_barrier();
    asm volatile("s_wait_dscnt 0" ::: "memory");
    __builtin_amdgcn_wave_barrier();
}

// ---------------- fragment loads ----------------
// A-fragment (16x32 bf16, MxK): lane l<16 holds row M=l, K = k0+0..7 and k0+16..23;
// lane l>=16 holds row M=l-16, K = k0+8..15 and k0+24..31.
// Caller passes rowptr for row (lane&15) and kbase = k0 + ((lane&16)?8:0).
__device__ __forceinline__ v16bf load_a16(const __bf16* rowptr, int kbase) {
    union { u32x4 u[2]; v16bf v; } t;
    t.u[0] = *(const u32x4*)(rowptr + kbase);
    t.u[1] = *(const u32x4*)(rowptr + kbase + 16);
    return t.v;
}

// B fragments are pre-packed so each lane reads 32 contiguous bytes.
template<int KST>
__device__ __forceinline__ void wmma_mm(v8f acc[8], const v16bf* afr, const __bf16* pB, int lane) {
#pragma unroll
    for (int nt = 0; nt < 8; ++nt) {
        v8f c = {};
#pragma unroll
        for (int ks = 0; ks < KST; ++ks) {
            v16bf b = *(const v16bf*)(pB + (((nt * KST + ks) * 32 + lane) << 4));
            c = __builtin_amdgcn_wmma_f32_16x16x32_bf16(false, afr[ks], false, b,
                                                        (short)0, c, false, false);
        }
        acc[nt] = c;
    }
}

// GroupNorm(num_groups=1) over 128 channels, on D-fragment registers.
// D layout: VGPR r, lanes 0-15 -> (M=r, N=lane); lanes 16-31 -> (M=8+r, N=lane-16).
__device__ __forceinline__ void groupnorm8(v8f acc[8], int lane,
                                           const float* g, const float* b, bool do_relu) {
    int col0 = lane & 15;
    float gv[8], bv[8];
#pragma unroll
    for (int nt = 0; nt < 8; ++nt) { gv[nt] = g[nt * 16 + col0]; bv[nt] = b[nt * 16 + col0]; }
#pragma unroll
    for (int r = 0; r < 8; ++r) {
        float s = 0.f, s2 = 0.f;
#pragma unroll
        for (int nt = 0; nt < 8; ++nt) { float x = acc[nt][r]; s += x; s2 += x * x; }
#pragma unroll
        for (int m = 1; m < 16; m <<= 1) { s += __shfl_xor(s, m); s2 += __shfl_xor(s2, m); }
        float mean = s * (1.0f / 128.0f);
        float var  = s2 * (1.0f / 128.0f) - mean * mean;
        float inv  = rsqrtf(var + GN_EPS);
#pragma unroll
        for (int nt = 0; nt < 8; ++nt) {
            float x = (acc[nt][r] - mean) * inv * gv[nt] + bv[nt];
            acc[nt][r] = do_relu ? fmaxf(x, 0.f) : x;
        }
    }
}

// Store D fragment to LDS as row-major bf16 [16][stride] (for restaging as A).
__device__ __forceinline__ void store_d_lds(const v8f acc[8], __bf16* lds, int stride,
                                            int lane, int colofs) {
    int rbase = (lane & 16) ? 8 : 0;
    int col0  = (lane & 15) + colofs;
#pragma unroll
    for (int r = 0; r < 8; ++r)
#pragma unroll
        for (int nt = 0; nt < 8; ++nt)
            lds[(rbase + r) * stride + nt * 16 + col0] = (__bf16)acc[nt][r];
}

// ---------------- utility kernels ----------------
__global__ void cvt_bf16_kernel(const float* __restrict__ x, __bf16* __restrict__ y, int n) {
    int i = blockIdx.x * blockDim.x + threadIdx.x;
    if (i < n) y[i] = (__bf16)x[i];
}

// Pack W [N=128][K] (row-major fp32, used as X @ W^T) into B-fragment order:
// out[((nt*ksteps+ks)*32+lane)*16 + j] = W[n][k], n = nt*16+(lane&15),
// k = ks*32 + ((lane&16)?8:0) + (j&7) + ((j&8)?16:0).
__global__ void pack_w_kernel(const float* __restrict__ W, __bf16* __restrict__ out, int K) {
    int ksteps = K >> 5;
    int total  = 8 * ksteps * 512;
    int idx = blockIdx.x * blockDim.x + threadIdx.x;
    if (idx >= total) return;
    int j    = idx & 15;
    int lane = (idx >> 4) & 31;
    int ks   = (idx >> 9) % ksteps;
    int nt   = (idx >> 9) / ksteps;
    int n = nt * 16 + (lane & 15);
    int k = ks * 32 + ((lane & 16) ? 8 : 0) + (j & 7) + ((j & 8) ? 16 : 0);
    out[idx] = (__bf16)W[n * K + k];
}

// ---------------- node GEMM: Y = Xbf16 @ W^T (K=128), fp32 out ----------------
__global__ void node_gemm_kernel(const __bf16* __restrict__ X, const __bf16* __restrict__ pB,
                                 float* __restrict__ Y, int ntiles) {
    int lane = threadIdx.x & 31;
    int tile = blockIdx.x * (blockDim.x >> 5) + (threadIdx.x >> 5);
    if (tile >= ntiles) return;
    int el  = lane & 15;
    int off = (lane & 16) ? 8 : 0;
    const __bf16* rowptr = X + (size_t)(tile * 16 + el) * DCH;
    v16bf afr[4];
#pragma unroll
    for (int ks = 0; ks < 4; ++ks) afr[ks] = load_a16(rowptr, ks * 32 + off);
    v8f acc[8];
    wmma_mm<4>(acc, afr, pB, lane);
    int rbase = tile * 16 + ((lane & 16) ? 8 : 0);
#pragma unroll
    for (int r = 0; r < 8; ++r)
#pragma unroll
        for (int nt = 0; nt < 8; ++nt)
            Y[(size_t)(rbase + r) * DCH + nt * 16 + el] = acc[nt][r];
}

// ---------------- fused edge kernel: 16 edges per wave ----------------
#define EW 4
__global__ void edge_kernel(const float* __restrict__ agt_ctrs, const float* __restrict__ ctx_ctrs,
                            const __bf16* __restrict__ actbf, const __bf16* __restrict__ ctxbf,
                            const int* __restrict__ hi, const int* __restrict__ wi,
                            const float* __restrict__ dw1, const float* __restrict__ db1,
                            const __bf16* __restrict__ pDist, const float* __restrict__ dg,
                            const float* __restrict__ db,
                            const __bf16* __restrict__ pQuery, const float* __restrict__ qg,
                            const float* __restrict__ qb,
                            const __bf16* __restrict__ pCtx1, const float* __restrict__ cg,
                            const float* __restrict__ cb,
                            const __bf16* __restrict__ pCtx2,
                            float* __restrict__ aacc, int E, int ntiles) {
    __shared__ __align__(32) __bf16 s0[EW][16 * DCH];   // 4KB/wave: h1 / gemm3-out restage
    __shared__ __align__(32) __bf16 cc[EW][16 * 256];   // 8KB/wave: [dist | query]
    int lane = threadIdx.x & 31;
    int wv   = threadIdx.x >> 5;
    int tile = blockIdx.x * EW + wv;
    if (tile >= ntiles) return;
    __bf16* S0 = s0[wv];
    __bf16* CC = cc[wv];

    int e0   = tile * 16;
    int el   = lane & 15;
    int half = lane >> 4;
    int off  = half ? 8 : 0;
    int ee   = e0 + el; if (ee > E - 1) ee = E - 1;
    int ha   = hi[ee];
    int wb   = wi[ee];

    // --- h1 = relu(d @ dist_w1^T + b1), K=2 done in VALU; stage bf16 to S0 ---
    float dx = agt_ctrs[2 * ha]     - ctx_ctrs[2 * wb];
    float dy = agt_ctrs[2 * ha + 1] - ctx_ctrs[2 * wb + 1];
    int nb0 = half * 64;
#pragma unroll 8
    for (int j = 0; j < 64; ++j) {
        int n = nb0 + j;
        float v = fmaxf(dw1[2 * n] * dx + dw1[2 * n + 1] * dy + db1[n], 0.f);
        S0[el * DCH + n] = (__bf16)v;
    }
    lds_fence_wave();

    v8f acc[8];
    {   // --- GEMM1: h1 @ dist_w2^T ---
        v16bf afr[4];
        const __bf16* rp = S0 + el * DCH;
#pragma unroll
        for (int ks = 0; ks < 4; ++ks) afr[ks] = load_a16(rp, ks * 32 + off);
        wmma_mm<4>(acc, afr, pDist, lane);
    }
    groupnorm8(acc, lane, dg, db, true);
    store_d_lds(acc, CC, 256, lane, 0);        // cols 0..127 = d

    {   // --- GEMM2: agts[hi] @ query_w^T (A gathered straight from global) ---
        v16bf afr[4];
        const __bf16* rp = actbf + (size_t)ha * DCH;
#pragma unroll
        for (int ks = 0; ks < 4; ++ks) afr[ks] = load_a16(rp, ks * 32 + off);
        wmma_mm<4>(acc, afr, pQuery, lane);
    }
    groupnorm8(acc, lane, qg, qb, true);
    store_d_lds(acc, CC, 256, lane, 128);      // cols 128..255 = q
    lds_fence_wave();

    {   // --- GEMM3: concat([d,q,ctx[wi]]) @ ctx_w1^T, K=384 ---
        v16bf afr[12];
        const __bf16* rp0 = CC + el * 256;
        const __bf16* rp1 = ctxbf + (size_t)wb * DCH;
#pragma unroll
        for (int ks = 0; ks < 8; ++ks)  afr[ks] = load_a16(rp0, ks * 32 + off);
#pragma unroll
        for (int ks = 8; ks < 12; ++ks) afr[ks] = load_a16(rp1, (ks - 8) * 32 + off);
        wmma_mm<12>(acc, afr, pCtx1, lane);
    }
    groupnorm8(acc, lane, cg, cb, true);
    store_d_lds(acc, S0, DCH, lane, 0);        // restage for GEMM4
    lds_fence_wave();

    {   // --- GEMM4: @ ctx_w2^T (no activation) ---
        v16bf afr[4];
        const __bf16* rp = S0 + el * DCH;
#pragma unroll
        for (int ks = 0; ks < 4; ++ks) afr[ks] = load_a16(rp, ks * 32 + off);
        wmma_mm<4>(acc, afr, pCtx2, lane);
    }

    // --- scatter: a.at[hi].add(c2) ---
    int rbase = half * 8;
#pragma unroll
    for (int r = 0; r < 8; ++r) {
        int em = e0 + rbase + r;
        if (em < E) {
            int hrow = hi[em];
#pragma unroll
            for (int nt = 0; nt < 8; ++nt)
                atomicAdd(&aacc[(size_t)hrow * DCH + nt * 16 + el], acc[nt][r]);
        }
    }
}

// ---------------- finalize: relu(gn(a)) -> @lin_w^T -> gn -> relu(+res) ----------------
#define FW 8
__global__ void finalize_kernel(const float* __restrict__ A, const float* __restrict__ res,
                                const __bf16* __restrict__ pLin,
                                const float* __restrict__ ng, const float* __restrict__ nb,
                                const float* __restrict__ lg, const float* __restrict__ lb,
                                float* __restrict__ out, int ntiles) {
    __shared__ __align__(32) __bf16 s0[FW][16 * DCH];
    int lane = threadIdx.x & 31;
    int wv   = threadIdx.x >> 5;
    int tile = blockIdx.x * FW + wv;
    if (tile >= ntiles) return;
    __bf16* S0 = s0[wv];

    int r   = lane & 15;
    int row = tile * 16 + r;
    int cb  = (lane >> 4) * 64;
    const float* ap = A + (size_t)row * DCH + cb;
    float s = 0.f, s2 = 0.f;
#pragma unroll 8
    for (int j = 0; j < 64; ++j) { float x = ap[j]; s += x; s2 += x * x; }
    s  += __shfl_xor(s, 16);
    s2 += __shfl_xor(s2, 16);
    float mean = s * (1.0f / 128.0f);
    float var  = s2 * (1.0f / 128.0f) - mean * mean;
    float inv  = rsqrtf(var + GN_EPS);
#pragma unroll 8
    for (int j = 0; j < 64; ++j) {
        int col = cb + j;
        float x = fmaxf((ap[j] - mean) * inv * ng[col] + nb[col], 0.f);
        S0[r * DCH + col] = (__bf16)x;
    }
    lds_fence_wave();

    int el  = lane & 15;
    int off = (lane & 16) ? 8 : 0;
    v16bf afr[4];
    const __bf16* rp = S0 + el * DCH;
#pragma unroll
    for (int ks = 0; ks < 4; ++ks) afr[ks] = load_a16(rp, ks * 32 + off);
    v8f acc[8];
    wmma_mm<4>(acc, afr, pLin, lane);
    groupnorm8(acc, lane, lg, lb, false);

    int rbase = tile * 16 + ((lane & 16) ? 8 : 0);
#pragma unroll
    for (int rr = 0; rr < 8; ++rr)
#pragma unroll
        for (int nt = 0; nt < 8; ++nt) {
            size_t idx = (size_t)(rbase + rr) * DCH + nt * 16 + el;
            out[idx] = fmaxf(acc[nt][rr] + res[idx], 0.f);
        }
}

// ---------------- host launcher ----------------
extern "C" void kernel_launch(void* const* d_in, const int* in_sizes, int n_in,
                              void* d_out, int out_size, void* d_ws, size_t ws_size,
                              hipStream_t stream) {
    const float* actors   = (const float*)d_in[0];
    const float* agt_ctrs = (const float*)d_in[1];
    const float* ctx      = (const float*)d_in[2];
    const float* ctx_ctrs = (const float*)d_in[3];
    const int*   hi       = (const int*)d_in[40];
    const int*   wi       = (const int*)d_in[41];
    int n_agt = in_sizes[0] / DCH;
    int n_ctx = in_sizes[2] / DCH;
    int E     = in_sizes[40];

    // workspace carve-out (all 256B aligned)
    size_t off = 0;
    auto alloc = [&](size_t bytes) -> void* {
        void* p = (char*)d_ws + off;
        off += (bytes + 255) & ~(size_t)255;
        return p;
    };
    __bf16* ctxbf = (__bf16*)alloc((size_t)n_ctx * DCH * 2);
    __bf16* actbf = (__bf16*)alloc((size_t)n_agt * DCH * 2);
    float*  aacc  = (float*)alloc((size_t)n_agt * DCH * 4);
    float*  mid   = (float*)alloc((size_t)n_agt * DCH * 4);
    __bf16* pw[2][6];
    static const int pK[6] = {128, 128, 384, 128, 128, 128}; // dist_w2,query_w,ctx_w1,ctx_w2,agt_w,lin_w
    for (int L = 0; L < 2; ++L)
        for (int i = 0; i < 6; ++i)
            pw[L][i] = (__bf16*)alloc((size_t)DCH * pK[i] * 2);

    auto pf = [&](int L, int i) { return (const float*)d_in[4 + L * 18 + i]; };
    // param index map: 0 dist_w1, 1 dist_b1, 2 dist_w2, 3 dist_g, 4 dist_b,
    // 5 query_w, 6 query_g, 7 query_b, 8 ctx_w1, 9 ctx_g, 10 ctx_b, 11 ctx_w2,
    // 12 agt_w, 13 norm_g, 14 norm_b, 15 lin_w, 16 lin_g, 17 lin_b
    static const int widx[6] = {2, 5, 8, 11, 12, 15};

    // ctx -> bf16 (once)
    {
        int n = n_ctx * DCH;
        cvt_bf16_kernel<<<(n + 255) / 256, 256, 0, stream>>>(ctx, ctxbf, n);
    }
    // pack all weights
    for (int L = 0; L < 2; ++L)
        for (int i = 0; i < 6; ++i) {
            int total = 8 * (pK[i] >> 5) * 512;
            pack_w_kernel<<<(total + 255) / 256, 256, 0, stream>>>(pf(L, widx[i]), pw[L][i], pK[i]);
        }

    int ntilesA = n_agt / 16;
    int ntilesE = (E + 15) / 16;

    for (int L = 0; L < 2; ++L) {
        const float* ain  = (L == 0) ? actors : mid;
        float*       aout = (L == 1) ? (float*)d_out : mid;

        {   // actors -> bf16
            int n = n_agt * DCH;
            cvt_bf16_kernel<<<(n + 255) / 256, 256, 0, stream>>>(ain, actbf, n);
        }
        // a = agts @ agt_w^T
        node_gemm_kernel<<<(ntilesA + 7) / 8, 256, 0, stream>>>(actbf, pw[L][4], aacc, ntilesA);
        // fused edge pipeline + scatter-add
        edge_kernel<<<(ntilesE + EW - 1) / EW, 32 * EW, 0, stream>>>(
            agt_ctrs, ctx_ctrs, actbf, ctxbf, hi, wi,
            pf(L, 0), pf(L, 1),
            pw[L][0], pf(L, 3), pf(L, 4),
            pw[L][1], pf(L, 6), pf(L, 7),
            pw[L][2], pf(L, 9), pf(L, 10),
            pw[L][3],
            aacc, E, ntilesE);
        // relu(gn(a)) -> @lin_w^T -> gn -> relu(+res)
        finalize_kernel<<<(ntilesA + FW - 1) / FW, 32 * FW, 0, stream>>>(
            aacc, ain, pw[L][5], pf(L, 13), pf(L, 14), pf(L, 16), pf(L, 17), aout, ntilesA);
    }
}
